// MultiHeadAttention_37177236914644
// MI455X (gfx1250) — matrix operand order repaired and checked by reading, not compiled, not run on hardware
//
#include <hip/hip_runtime.h>
#include <math.h>

// Problem constants from the reference
#define B_  4
#define S_  2048
#define H_  16
#define D_  64
#define DM_ 1024

typedef __attribute__((ext_vector_type(16))) __bf16        v16bf;
typedef __attribute__((ext_vector_type(8)))  float         v8f;
typedef __attribute__((ext_vector_type(4)))  unsigned int  v4u;
typedef __attribute__((ext_vector_type(8)))  int           v8i_t;
typedef __attribute__((ext_vector_type(4)))  int           v4i_t;

union Frag { v16bf v; unsigned short u[16]; uint4 q[2]; };
union Acc  { v8f   v; float f[8]; };

#if defined(__has_builtin)
#if __has_builtin(__builtin_amdgcn_tensor_load_to_lds)
#define HAVE_TDM 1
#endif
#endif

// float -> bf16 via native conversion (lets the compiler use HW cvt ops)
__device__ __forceinline__ unsigned short f2bf(float x) {
    return __builtin_bit_cast(unsigned short, (__bf16)x);
}

// ---------------------------------------------------------------------------
// Projection GEMM: y = X @ W + bias, output bf16.
//   TR=false: y stored head-split [B,H,S,D]  (for Q, K)
//   TR=true : y stored transposed [B,H,D,S]  (for V -> PV B-operand layout)
// Block: 128 threads (4 waves). Block tile: 128(M) x 64(N), K-step 32.
// Each wave computes a 32x64 strip: 2 A-frags x 4 B-frags = 8 WMMAs / k-step.
// ---------------------------------------------------------------------------
#define LDA 40   // 32 + 8 pad (ushorts); 80B row stride, 16B aligned
#define LDB 40

template <bool TR>
__global__ __launch_bounds__(128)
void proj_gemm_bf16(const float* __restrict__ X, const float* __restrict__ W,
                    const float* __restrict__ bias, unsigned short* __restrict__ out)
{
    __shared__ __align__(16) unsigned short As[128 * LDA];  // A tile [m][k] bf16
    __shared__ __align__(16) unsigned short Bs[64 * LDB];   // B tile [n][k] bf16 (transposed)

    const int t    = threadIdx.x;
    const int lane = t & 31;
    const int wave = t >> 5;              // m-subtile 0..3 (32 rows each)
    const int ln   = lane & 15;
    const int lh   = lane >> 4;           // half-wave select
    const int m0   = blockIdx.x * 128;
    const int n0   = blockIdx.y * 64;

    Acc acc[2][4];
    #pragma unroll
    for (int r = 0; r < 2; r++)
        #pragma unroll
        for (int i = 0; i < 4; i++)
            #pragma unroll
            for (int j = 0; j < 8; j++) acc[r][i].f[j] = 0.0f;

    for (int k0 = 0; k0 < DM_; k0 += 32) {
        // ---- stage A tile: 128 rows x 32 cols fp32 -> bf16 LDS (1 row / thread)
        {
            const float* src = X + (size_t)(m0 + t) * DM_ + k0;
            unsigned short* dst = &As[t * LDA];
            #pragma unroll
            for (int i = 0; i < 32; i += 4) {
                float4 f = *(const float4*)(src + i);
                dst[i+0] = f2bf(f.x); dst[i+1] = f2bf(f.y);
                dst[i+2] = f2bf(f.z); dst[i+3] = f2bf(f.w);
            }
        }
        // ---- stage B tile transposed: W[k0+kk][n0+n] -> Bs[n][kk]
        {
            const int kk = t >> 2;
            const int c0 = (t & 3) * 16;
            const float* src = W + (size_t)(k0 + kk) * DM_ + n0 + c0;
            #pragma unroll
            for (int i = 0; i < 16; i += 4) {
                float4 f = *(const float4*)(src + i);
                Bs[(c0+i+0) * LDB + kk] = f2bf(f.x);
                Bs[(c0+i+1) * LDB + kk] = f2bf(f.y);
                Bs[(c0+i+2) * LDB + kk] = f2bf(f.z);
                Bs[(c0+i+3) * LDB + kk] = f2bf(f.w);
            }
        }
        __syncthreads();

        // ---- A fragments (16x32 bf16): lane holds row m; 16 contiguous K per half-wave
        Frag a[2];
        #pragma unroll
        for (int r = 0; r < 2; r++) {
            const int m  = wave * 32 + r * 16 + ln;
            const int kh = lh * 16;
            a[r].q[0] = *(const uint4*)&As[m * LDA + kh];
            a[r].q[1] = *(const uint4*)&As[m * LDA + kh + 8];
        }
        // ---- B fragments (32x16 bf16): lane holds col n; 16 contiguous K per half-wave
        #pragma unroll
        for (int nt = 0; nt < 4; nt++) {
            Frag b;
            const int n  = nt * 16 + ln;
            const int kb = lh * 16;
            b.q[0] = *(const uint4*)&Bs[n * LDB + kb];
            b.q[1] = *(const uint4*)&Bs[n * LDB + kb + 8];
            #pragma unroll
            for (int r = 0; r < 2; r++)
                acc[r][nt].v = __builtin_amdgcn_wmma_f32_16x16x32_bf16(
                    false, a[r].v, false, b.v, (short)0, acc[r][nt].v, false, false);
        }
        __syncthreads();
    }

    // ---- epilogue: bias add, bf16, scatter to [B,H,S,D] or [B,H,D,S]
    #pragma unroll
    for (int nt = 0; nt < 4; nt++) {
        const int col = n0 + nt * 16 + ln;
        const float bv = bias[col];
        const int h = col >> 6;
        const int d = col & 63;
        const int mh = lh * 8;
        #pragma unroll
        for (int r = 0; r < 2; r++) {
            #pragma unroll
            for (int j = 0; j < 8; j++) {
                const int row = m0 + wave * 32 + r * 16 + mh + j;  // m = b*S + s
                const int b = row >> 11;                           // /2048
                const int s = row & (S_ - 1);
                const unsigned short val = f2bf(acc[r][nt].f[j] + bv);
                if (TR)
                    out[((size_t)(b * H_ + h) * D_ + d) * S_ + s] = val;
                else
                    out[((size_t)(b * H_ + h) * S_ + s) * D_ + d] = val;
            }
        }
    }
}

// ---------------------------------------------------------------------------
// TDM: DMA a 64(d-rows) x 32(k) bf16 tile from transposed V [D,S] into LDS,
// with LDS padding: after every 16 DWORDs (one 64B row) insert 4 DWORDs (16B)
// -> LDS row stride 80B (16B aligned for ds_load_b128, 20-bank step).
// D# per ISA cdna5 §8.
// ---------------------------------------------------------------------------
#define VROW 40   // padded LDS row stride in ushorts (80B)

#ifdef HAVE_TDM
__device__ __forceinline__ void tdm_load_vt_tile(const unsigned short* gsrc,
                                                 unsigned short* ldst)
{
    const unsigned long long ga = (unsigned long long)(size_t)gsrc;
    v4u g0;
    g0[0] = 1u;                                    // count=1, flags 0
    g0[1] = (unsigned)(size_t)ldst;                // lds_addr (low 32 bits = LDS offset)
    g0[2] = (unsigned)(ga & 0xFFFFFFFFull);        // global_addr[31:0]
    g0[3] = (unsigned)((ga >> 32) & 0x01FFFFFFull) // global_addr[56:32]
          | (2u << 30);                            // type = 2 ("image")
    v8i_t g1;
    g1[0] = (int)(0x00010000u                      // data_size=1 (2 bytes)
          | (1u << 20)                             // pad_enable
          | (3u << 22)                             // pad_interval: 16 DWORDs
          | (3u << 25));                           // pad_amount:   4 DWORDs
    g1[1] = (int)(2048u << 16);    // tensor_dim0 = 2048 (row length, k)
    g1[2] = (int)(64u << 16);      // tensor_dim1 = 64 (d rows)
    g1[3] = (int)(32u << 16);      // tile_dim0 = 32 (k slice)
    g1[4] = 64;                    // tile_dim1 = 64 (d rows), tile_dim2 = 0
    g1[5] = 2048;                  // tensor_dim0_stride = S
    g1[6] = 0;
    g1[7] = 0;
    v4i_t z4 = {0, 0, 0, 0};
#if __clang_major__ >= 23
    v8i_t z8 = {0, 0, 0, 0, 0, 0, 0, 0};
    __builtin_amdgcn_tensor_load_to_lds(g0, g1, z4, z4, z8, 0);   // clang-23 form
#else
    __builtin_amdgcn_tensor_load_to_lds(g0, g1, z4, z4, 0);       // ROCm 7.2 form
#endif
}
#endif

// ---------------------------------------------------------------------------
// Attention: block = (q-tile of 16, head); 4 waves, wave b owns batch b.
// Softmax is over the batch axis (4 values, pointwise in (q,k)) via LDS exchange.
// V tiles stream in via double-buffered TDM (TENSORcnt-pipelined).
// ---------------------------------------------------------------------------
__global__ __launch_bounds__(128)
void attn_kernel(const unsigned short* __restrict__ Q,
                 const unsigned short* __restrict__ K,
                 const unsigned short* __restrict__ Vt,   // [B,H,D,S]
                 const float* __restrict__ mask,
                 float* __restrict__ out)
{
    __shared__ __align__(16) float          sc_s[B_][2][16][17];  // masked scores
    __shared__ __align__(16) unsigned short w_s[B_][16][32];      // softmax weights bf16 [q][k]
#ifdef HAVE_TDM
    __shared__ __align__(16) unsigned short Vs[2][B_][D_ * VROW]; // double-buffered V tiles
#endif

    const int t    = threadIdx.x;
    const int lane = t & 31;
    const int b    = t >> 5;          // wave index == batch index
    const int ln   = lane & 15;
    const int lh   = lane >> 4;
    const int qt   = blockIdx.x;      // q-tile (16 rows)
    const int h    = blockIdx.y;

    const size_t head = (size_t)(b * H_ + h) * S_ * D_;
    const unsigned short* Qb = Q  + head;
    const unsigned short* Kb = K  + head;
    const unsigned short* Vb = Vt + head;    // transposed layout, same head stride
    const float* maskb = mask + (size_t)b * S_ * S_;

    // Q A-fragments for the two 32-wide d-steps (row = q, K-dim = d)
    Frag aq[2];
    {
        const int s  = qt * 16 + ln;
        const int kh = lh * 16;
        const unsigned short* qrow = Qb + (size_t)s * D_;
        #pragma unroll
        for (int ds = 0; ds < 2; ds++) {
            aq[ds].q[0] = *(const uint4*)(qrow + ds * 32 + kh);
            aq[ds].q[1] = *(const uint4*)(qrow + ds * 32 + kh + 8);
        }
    }

    Acc co[4];                        // PV accumulators: 4 x (16q x 16d)
    #pragma unroll
    for (int i = 0; i < 4; i++)
        #pragma unroll
        for (int j = 0; j < 8; j++) co[i].f[j] = 0.0f;

    const int mh = lh * 8;            // C/D layout row base per half-wave

#ifdef HAVE_TDM
    tdm_load_vt_tile(Vb, &Vs[0][b][0]);     // prime the pipeline (k0 = 0)
    int cur = 0;
#endif

    for (int k0 = 0; k0 < S_; k0 += 32) {
#ifdef HAVE_TDM
        if (k0 + 32 < S_)             // async-issue next slab into the other buffer
            tdm_load_vt_tile(Vb + (k0 + 32), &Vs[cur ^ 1][b][0]);
#endif
        // ---- QK^T: two 16-wide score tiles, contraction over d=64 (2 WMMA steps)
        Acc sc[2];
        #pragma unroll
        for (int j = 0; j < 2; j++) {
            const int kcol = k0 + j * 16 + ln;        // B-frag: lane holds column (k)
            const unsigned short* krow = Kb + (size_t)kcol * D_;
            const int kb = lh * 16;
            Acc c;
            #pragma unroll
            for (int x = 0; x < 8; x++) c.f[x] = 0.0f;
            #pragma unroll
            for (int ds = 0; ds < 2; ds++) {
                Frag bk;
                bk.q[0] = *(const uint4*)(krow + ds * 32 + kb);
                bk.q[1] = *(const uint4*)(krow + ds * 32 + kb + 8);
                c.v = __builtin_amdgcn_wmma_f32_16x16x32_bf16(
                    false, aq[ds].v, false, bk.v, (short)0, c.v, false, false);
            }
            sc[j] = c;
        }

        // ---- scale + mask, publish to LDS
        #pragma unroll
        for (int j = 0; j < 2; j++) {
            const int kc = k0 + j * 16 + ln;
            #pragma unroll
            for (int jj = 0; jj < 8; jj++) {
                const int qrow = qt * 16 + mh + jj;
                const float m = maskb[(size_t)qrow * S_ + kc];
                sc_s[b][j][mh + jj][ln] = sc[j].f[jj] * 0.125f + m * (-1.0e9f);
            }
        }
        __syncthreads();

        // ---- softmax over batch axis (4 values), keep own batch's weight (bf16)
        #pragma unroll
        for (int j = 0; j < 2; j++) {
            #pragma unroll
            for (int jj = 0; jj < 8; jj++) {
                const int m = mh + jj;
                const float v0 = sc_s[0][j][m][ln];
                const float v1 = sc_s[1][j][m][ln];
                const float v2 = sc_s[2][j][m][ln];
                const float v3 = sc_s[3][j][m][ln];
                const float mx = fmaxf(fmaxf(v0, v1), fmaxf(v2, v3));
                const float e0 = __expf(v0 - mx), e1 = __expf(v1 - mx);
                const float e2 = __expf(v2 - mx), e3 = __expf(v3 - mx);
                const float denom = e0 + e1 + e2 + e3;
                const float mine = (b == 0) ? e0 : (b == 1) ? e1 : (b == 2) ? e2 : e3;
                w_s[b][m][j * 16 + ln] = f2bf(mine / denom);
            }
        }
        __syncthreads();              // protect sc_s before next iteration overwrites

        // ---- weights -> A fragment (16q x 32k) via LDS transpose (wave-local region)
        Frag aw;
        {
            const int kh = lh * 16;
            aw.q[0] = *(const uint4*)&w_s[b][ln][kh];
            aw.q[1] = *(const uint4*)&w_s[b][ln][kh + 8];
        }

#ifdef HAVE_TDM
        if (k0 + 32 < S_) __builtin_amdgcn_s_wait_tensorcnt(1);  // current buffer ready
        else              __builtin_amdgcn_s_wait_tensorcnt(0);
        asm volatile("" ::: "memory");
#endif

        // ---- PV: out[q, d] += w[q, k32] @ V[k32, d], 4 d-tiles of 16.
        // B-frag: lane holds column d -> 16 contiguous k from transposed V row.
        #pragma unroll
        for (int dt = 0; dt < 4; dt++) {
            Frag bv;
            const int d  = dt * 16 + ln;
            const int kb = lh * 16;
#ifdef HAVE_TDM
            const unsigned short* vrow = &Vs[cur][b][d * VROW];
            bv.q[0] = *(const uint4*)(vrow + kb);
            bv.q[1] = *(const uint4*)(vrow + kb + 8);
#else
            const unsigned short* vrow = Vb + (size_t)d * S_ + k0;
            bv.q[0] = *(const uint4*)(vrow + kb);
            bv.q[1] = *(const uint4*)(vrow + kb + 8);
#endif
            co[dt].v = __builtin_amdgcn_wmma_f32_16x16x32_bf16(
                false, aw.v, false, bv.v, (short)0, co[dt].v, false, false);
        }
#ifdef HAVE_TDM
        cur ^= 1;
#endif
        // gfx1250 prefetch of next K slab
        if (k0 + 32 < S_)
            __builtin_prefetch((const void*)(Kb + (size_t)(k0 + 32 + ln) * D_), 0, 1);
    }

    // ---- write output: [B, S, H*D] fp32
    #pragma unroll
    for (int dt = 0; dt < 4; dt++) {
        const int d = dt * 16 + ln;
        #pragma unroll
        for (int jj = 0; jj < 8; jj++) {
            const int s = qt * 16 + mh + jj;
            out[((size_t)(b * S_ + s)) * DM_ + h * D_ + d] = co[dt].f[jj];
        }
    }
}

// ---------------------------------------------------------------------------
extern "C" void kernel_launch(void* const* d_in, const int* in_sizes, int n_in,
                              void* d_out, int out_size, void* d_ws, size_t ws_size,
                              hipStream_t stream)
{
    (void)in_sizes; (void)n_in; (void)out_size; (void)ws_size;

    const float* input_q = (const float*)d_in[0];
    const float* input_k = (const float*)d_in[1];
    const float* input_v = (const float*)d_in[2];
    const float* mask    = (const float*)d_in[3];
    const float* Wq      = (const float*)d_in[4];
    const float* bq      = (const float*)d_in[5];
    const float* Wk      = (const float*)d_in[6];
    const float* bk      = (const float*)d_in[7];
    const float* Wv      = (const float*)d_in[8];
    const float* bv      = (const float*)d_in[9];

    // workspace: Q,K head-split [B,H,S,D], V transposed [B,H,D,S]; bf16, 48 MB
    unsigned short* qkv = (unsigned short*)d_ws;
    const size_t QS = (size_t)B_ * H_ * S_ * D_;
    unsigned short* Qw  = qkv;
    unsigned short* Kw  = qkv + QS;
    unsigned short* Vtw = qkv + 2 * QS;

    dim3 pgrid(B_ * S_ / 128, DM_ / 64, 1);  // 64 x 16 blocks
    proj_gemm_bf16<false><<<pgrid, 128, 0, stream>>>(input_q, Wq, bq, Qw);
    proj_gemm_bf16<false><<<pgrid, 128, 0, stream>>>(input_k, Wk, bk, Kw);
    proj_gemm_bf16<true ><<<pgrid, 128, 0, stream>>>(input_v, Wv, bv, Vtw);

    dim3 agrid(S_ / 16, H_, 1);              // 128 x 16 blocks, 4 waves each
    attn_kernel<<<agrid, 128, 0, stream>>>(Qw, Kw, Vtw, mask, (float*)d_out);
}
